// Mamba_block_32006096290423
// MI455X (gfx1250) — matrix operand order repaired and checked by reading, not compiled, not run on hardware
//
#include <hip/hip_runtime.h>
#include <hip/hip_bf16.h>
#include <math.h>

// ---------------------------------------------------------------------------
// CDNA5 (gfx1250) wave32 WMMA helpers.
// Layouts per CDNA5 ISA 7.12.2:
//   A (16x32 f16): lane l -> row = l&15, g = l>>4.
//     element e: K = 16*g+e (contiguous order used here, not the ISA table order)
//   B (32x16 f16): lane l -> col = l&15, g = l>>4. element e: K = 16*g+e (same order as A; this code's choice)
//   C/D (16x16 f32, 8 VGPRs): lane l -> col = l&15; reg r -> row = r + 8*(l>>4)
// ---------------------------------------------------------------------------
typedef __attribute__((ext_vector_type(16))) _Float16 v16h;
typedef __attribute__((ext_vector_type(8)))  float    v8f;

union AFrag { v16h v; _Float16 e[16]; };
union CFrag { v8f  v; float    e[8];  };

__device__ __forceinline__ int wmma_a_koff(int e, int g) {
    return 16 * g + e;   // contiguous: K = 16g + e
}

// direction l-map for SS2D (involution for k=1,2,3; identity for k=0)
__device__ __forceinline__ int dir_map(int k, int l) {
    switch (k & 3) {
        case 0: return l;
        case 1: return ((l & 31) << 5) | (l >> 5);
        case 2: return 1023 - l;
        default: { int t = 1023 - l; return ((t & 31) << 5) | (t >> 5); }
    }
}

// LDS byte offset of a __shared__ object = low 32 bits of its generic address
// (LDS aperture, CDNA5 ISA 10.2).
__device__ __forceinline__ unsigned lds_off_of(const void* p) {
    return (unsigned)(size_t)p;
}

// async global->LDS 16B copy, tracked by ASYNCcnt (CDNA5 ISA 10 / 15.18.3)
__device__ __forceinline__ void async_copy_b128(unsigned lds_off, const float* gptr) {
    unsigned long long ga = (unsigned long long)gptr;
    asm volatile("global_load_async_to_lds_b128 %0, %1, off"
                 :: "v"(lds_off), "v"(ga) : "memory");
}

// ---------------------------------------------------------------------------
// Register-blocked WMMA GEMM: C[M x N] = A[M x K] * W[K x N] (f32 memory,
// f16 fragments, f32 accumulate).
// Block = 256 threads (8 waves, 2x4 wave grid) -> 128x128 output tile.
// Each wave owns 64x32 (4 M-subtiles x 2 N-subtiles) = 8 WMMAs / K-step.
// A (128x32 f32) and B (32x128 f32) tiles double-buffered in LDS, filled by
// global_load_async_to_lds_b128 and retired with s_wait_asynccnt.
// M % 128 == 0, N % 128 == 0, K % 32 == 0 for every use below.
// ---------------------------------------------------------------------------
__global__ __launch_bounds__(256)
void gemm_wmma_kernel(const float* __restrict__ A,
                      const float* __restrict__ W,
                      float* __restrict__ C,
                      int K, int lda, int ldb, int ldc) {
    __shared__ float As[2][128 * 32];   // [buf][m][k]  16 KB each
    __shared__ float Bs[2][32 * 128];   // [buf][k][n]  16 KB each

    const int tid  = threadIdx.x;
    const int lane = tid & 31;
    const int col  = lane & 15;
    const int grp  = lane >> 4;
    const int wave = tid >> 5;
    const int wm   = wave & 1;          // 0..1 : 64-row group
    const int wn   = wave >> 1;         // 0..3 : 32-col group
    const int m0   = blockIdx.x * 128;
    const int n0   = blockIdx.y * 128;

    const int nsteps = K >> 5;

    // issue one K-tile's async copies into buffer `buf`
    auto issue = [&](int k0, int buf) {
        #pragma unroll
        for (int j = 0; j < 4; ++j) {                 // A tile: 4096 floats
            int f = (tid + j * 256) * 4;
            int rm = f >> 5, kk = f & 31;
            async_copy_b128(lds_off_of(&As[buf][f]),
                            &A[(size_t)(m0 + rm) * lda + k0 + kk]);
        }
        #pragma unroll
        for (int j = 0; j < 4; ++j) {                 // B tile: 4096 floats
            int f = (tid + j * 256) * 4;
            int rk = f >> 7, nn = f & 127;
            async_copy_b128(lds_off_of(&Bs[buf][f]),
                            &W[(size_t)(k0 + rk) * ldb + n0 + nn]);
        }
    };

    CFrag acc[4][2] = {};

    issue(0, 0);
    for (int it = 0; it < nsteps; ++it) {
        const int cur = it & 1;
        const bool has_next = (it + 1) < nsteps;
        if (has_next) issue((it + 1) << 5, 1 - cur);  // overlap next tile

        // retire the older 8-instruction group (async loads complete in order)
        if (has_next) asm volatile("s_wait_asynccnt 8" ::: "memory");
        else          asm volatile("s_wait_asynccnt 0" ::: "memory");
        __syncthreads();                               // cross-wave visibility

        AFrag a[4], b[2];
        #pragma unroll
        for (int mt = 0; mt < 4; ++mt) {
            const int r = wm * 64 + mt * 16 + col;
            #pragma unroll
            for (int e = 0; e < 16; ++e)
                a[mt].e[e] = (_Float16)As[cur][r * 32 + wmma_a_koff(e, grp)];
        }
        #pragma unroll
        for (int nt = 0; nt < 2; ++nt) {
            const int cn = wn * 32 + nt * 16 + col;
            #pragma unroll
            for (int e = 0; e < 16; ++e)
                b[nt].e[e] = (_Float16)Bs[cur][((grp << 4) + e) * 128 + cn];
        }
        #pragma unroll
        for (int mt = 0; mt < 4; ++mt)
            #pragma unroll
            for (int nt = 0; nt < 2; ++nt)
                acc[mt][nt].v = __builtin_amdgcn_wmma_f32_16x16x32_f16(
                    false, a[mt].v, false, b[nt].v, (short)0,
                    acc[mt][nt].v, false, false);
        __syncthreads();                // buffer may be re-filled next iter
    }

    #pragma unroll
    for (int mt = 0; mt < 4; ++mt)
        #pragma unroll
        for (int nt = 0; nt < 2; ++nt)
            #pragma unroll
            for (int r = 0; r < 8; ++r)
                C[(size_t)(m0 + wm * 64 + mt * 16 + r + (grp << 3)) * ldc
                  + n0 + wn * 32 + nt * 16 + col] = acc[mt][nt].e[r];
}

// ---------------------------------------------------------------------------
// x_dbl = einsum('bkdl,kcd->bkcl').  M = l (1024), N = c (48), K = d (512).
// Direction remap folded into the A-fragment gather (xs never materialized).
// grid = (64 mtiles, 3 ntiles, 32 bk), block = 32 (one wave per 16x16 tile).
// ---------------------------------------------------------------------------
__global__ void xdbl_wmma_kernel(const float* __restrict__ xc,
                                 const float* __restrict__ xpw,
                                 float* __restrict__ xdbl) {
    const int lane = threadIdx.x & 31;
    const int col  = lane & 15;
    const int grp  = lane >> 4;
    const int m0   = blockIdx.x * 16;
    const int n0   = blockIdx.y * 16;
    const int bk   = blockIdx.z;
    const int b    = bk >> 2, k = bk & 3;
    const float* xcb = xc  + (size_t)b * 512 * 1024;
    const float* wk  = xpw + (size_t)k * 48 * 512;
    const int lsrc = dir_map(k, m0 + col);

    CFrag acc = {};
    for (int k0 = 0; k0 < 512; k0 += 32) {
        AFrag a, bf;
        #pragma unroll
        for (int e = 0; e < 16; ++e) {
            a.e[e]  = (_Float16)xcb[(size_t)(k0 + wmma_a_koff(e, grp)) * 1024 + lsrc];
            bf.e[e] = (_Float16)wk[(size_t)(n0 + col) * 512 + k0 + (grp << 4) + e];
        }
        acc.v = __builtin_amdgcn_wmma_f32_16x16x32_f16(
            false, a.v, false, bf.v, (short)0, acc.v, false, false);
    }
    #pragma unroll
    for (int r = 0; r < 8; ++r)
        xdbl[(size_t)(bk * 48 + n0 + col) * 1024 + m0 + r + (grp << 3)] = acc.e[r];
}

// ---------------------------------------------------------------------------
// Gram matrix per (b,head): G[32x32] = q(32x1024) * k^T(1024x32).
// q,k read from dwconv output (pixel-major, stride 768). grid=64, block=128.
// ---------------------------------------------------------------------------
__global__ void gram_wmma_kernel(const float* __restrict__ dwq,
                                 float* __restrict__ G) {
    const int lane = threadIdx.x & 31;
    const int wave = threadIdx.x >> 5;       // 0..3
    const int mt = wave >> 1, nt = wave & 1;
    const int col = lane & 15, grp = lane >> 4;
    const int bh = blockIdx.x;
    const int b = bh >> 3, head = bh & 7;
    const float* base = dwq + (size_t)b * 1024 * 768;

    CFrag acc = {};
    for (int k0 = 0; k0 < 1024; k0 += 32) {
        AFrag a, bf;
        #pragma unroll
        for (int e = 0; e < 16; ++e) {
            a.e[e]  = (_Float16)base[(size_t)(k0 + wmma_a_koff(e, grp)) * 768
                                     + head * 32 + mt * 16 + col];
            bf.e[e] = (_Float16)base[(size_t)(k0 + (grp << 4) + e) * 768
                                     + 256 + head * 32 + nt * 16 + col];
        }
        acc.v = __builtin_amdgcn_wmma_f32_16x16x32_f16(
            false, a.v, false, bf.v, (short)0, acc.v, false, false);
    }
    #pragma unroll
    for (int r = 0; r < 8; ++r)
        G[((size_t)bh * 32 + mt * 16 + r + (grp << 3)) * 32 + nt * 16 + col] = acc.e[r];
}

// ---------------------------------------------------------------------------
// out[c, l] = attn(32x32) @ v(32x1024) per (b,head). Single K=32 WMMA step.
// grid = (64 ntiles, 64 bh), block = 64 (2 waves = 2 m-tiles).
// ---------------------------------------------------------------------------
__global__ void attn_apply_wmma_kernel(const float* __restrict__ attn,
                                       const float* __restrict__ dwv,
                                       float* __restrict__ out) {
    const int lane = threadIdx.x & 31;
    const int wave = threadIdx.x >> 5;       // m tile 0/1
    const int col = lane & 15, grp = lane >> 4;
    const int n0 = blockIdx.x * 16;
    const int bh = blockIdx.y;
    const int b = bh >> 3, head = bh & 7;
    const float* ab = attn + (size_t)bh * 32 * 32;
    const float* vb = dwv + (size_t)b * 1024 * 768 + 512 + head * 32;

    AFrag a, bf;
    #pragma unroll
    for (int e = 0; e < 16; ++e) {
        a.e[e]  = (_Float16)ab[(wave * 16 + col) * 32 + wmma_a_koff(e, grp)];
        bf.e[e] = (_Float16)vb[(size_t)(n0 + col) * 768 + (grp << 4) + e];
    }
    CFrag acc = {};
    acc.v = __builtin_amdgcn_wmma_f32_16x16x32_f16(
        false, a.v, false, bf.v, (short)0, acc.v, false, false);
    #pragma unroll
    for (int r = 0; r < 8; ++r)
        out[(size_t)(b * 1024 + n0 + col) * 256 + head * 32 + wave * 16 + r + (grp << 3)]
            = acc.e[r];
}

// ----------------------------- elementwise --------------------------------
__global__ void zero_kernel(float* __restrict__ p, int n) {
    int i = blockIdx.x * blockDim.x + threadIdx.x;
    if (i < n) p[i] = 0.f;
}

__global__ void transpose_kernel(const float* __restrict__ W, float* __restrict__ Wt,
                                 int R, int Ccol) {
    int idx = blockIdx.x * blockDim.x + threadIdx.x;
    if (idx >= R * Ccol) return;
    int r = idx / Ccol, c = idx % Ccol;
    Wt[(size_t)c * R + r] = W[idx];
}

// mod = silu(t) @ W_ada + b_ada   (8 x 1024) @ (1024 x 1536)
__global__ void mod_kernel(const float* __restrict__ t, const float* __restrict__ Wada,
                           const float* __restrict__ bada, float* __restrict__ mod) {
    int idx = blockIdx.x * blockDim.x + threadIdx.x;
    if (idx >= 8 * 1536) return;
    int b = idx / 1536, j = idx % 1536;
    float s = bada[j];
    for (int kk = 0; kk < 1024; ++kk) {
        float tv = t[b * 1024 + kk];
        s += (tv / (1.f + __expf(-tv))) * Wada[(size_t)kk * 1536 + j];
    }
    mod[idx] = s;
}

// layernorm over C=256 + adaLN modulate. one 256-thread block per pixel.
__global__ void ln_mod_kernel(const float* __restrict__ X, int chan_major,
                              const float* __restrict__ g, const float* __restrict__ be,
                              const float* __restrict__ mod, int sh_chunk, int sc_chunk,
                              float eps, float* __restrict__ out) {
    __shared__ float red[256];
    int pix = blockIdx.x;
    int b = pix >> 10, l = pix & 1023;
    int c = threadIdx.x;
    float v = chan_major ? X[((size_t)(b * 256 + c)) * 1024 + l]
                         : X[(size_t)pix * 256 + c];
    red[c] = v; __syncthreads();
    for (int s = 128; s > 0; s >>= 1) { if (c < s) red[c] += red[c + s]; __syncthreads(); }
    float mu = red[0] * (1.f / 256.f); __syncthreads();
    float d = v - mu;
    red[c] = d * d; __syncthreads();
    for (int s = 128; s > 0; s >>= 1) { if (c < s) red[c] += red[c + s]; __syncthreads(); }
    float var = red[0] * (1.f / 256.f);
    float y = d * rsqrtf(var + eps);
    if (g) y = y * g[c] + be[c];
    float sh = mod[b * 1536 + sh_chunk * 256 + c];
    float sc = mod[b * 1536 + sc_chunk * 256 + c];
    out[(size_t)pix * 256 + c] = y * (1.f + sc) + sh;
}

// depthwise 3x3 SAME + bias + silu on xm (cols 0..511 of xm_z rows), NCHW out
__global__ void conv_silu_kernel(const float* __restrict__ xmz, const float* __restrict__ cw,
                                 const float* __restrict__ cb, float* __restrict__ xc) {
    int idx = blockIdx.x * blockDim.x + threadIdx.x;
    if (idx >= 8 * 512 * 1024) return;
    int l = idx & 1023, d = (idx >> 10) & 511, b = idx >> 19;
    int h = l >> 5, w = l & 31;
    float s = cb[d];
    #pragma unroll
    for (int dh = -1; dh <= 1; ++dh)
        #pragma unroll
        for (int dwi = -1; dwi <= 1; ++dwi) {
            int hh = h + dh, ww = w + dwi;
            if (hh < 0 || hh > 31 || ww < 0 || ww > 31) continue;
            s += xmz[((size_t)(b * 1024 + hh * 32 + ww)) * 1024 + d]
               * cw[d * 9 + (dh + 1) * 3 + (dwi + 1)];
        }
    xc[((size_t)(b * 512 + d)) * 1024 + l] = s / (1.f + __expf(-s));
}

// selective scan: block = 512 threads (d), grid = 32 (b*4+k); 16 states/thread.
// dt-projection + softplus fused; 4 directions scatter-accumulated (atomic).
__global__ void scan_kernel(const float* __restrict__ xc, const float* __restrict__ xdbl,
                            const float* __restrict__ dtw_all, const float* __restrict__ dtb_all,
                            const float* __restrict__ Alogs, const float* __restrict__ Ds,
                            float* __restrict__ ycomb) {
    __shared__ float sh[48];
    int bk = blockIdx.x; int b = bk >> 2, k = bk & 3;
    int d = threadIdx.x;
    float A[16], h[16], dtw[16];
    #pragma unroll
    for (int n = 0; n < 16; ++n) {
        A[n]   = -__expf(Alogs[(size_t)(k * 512 + d) * 16 + n]);
        dtw[n] = dtw_all[(size_t)(k * 512 + d) * 16 + n];
        h[n]   = 0.f;
    }
    float dtb = dtb_all[k * 512 + d];
    float Dv  = Ds[k * 512 + d];
    const float* xdb = xdbl + (size_t)bk * 48 * 1024;
    const float* xcb = xc + (size_t)b * 512 * 1024 + (size_t)d * 1024;
    float* yb = ycomb + (size_t)b * 512 * 1024 + (size_t)d * 1024;

    for (int l = 0; l < 1024; ++l) {
        if (threadIdx.x < 48) {
            sh[threadIdx.x] = xdb[(size_t)threadIdx.x * 1024 + l];
            if (l + 1 < 1024)                      // global_prefetch_b8
                __builtin_prefetch(&xdb[(size_t)threadIdx.x * 1024 + l + 1], 0, 1);
        }
        __syncthreads();
        float dtl = dtb;
        #pragma unroll
        for (int r = 0; r < 16; ++r) dtl += dtw[r] * sh[r];
        float dt = (dtl > 20.f) ? dtl : log1pf(__expf(dtl));
        int lsrc = dir_map(k, l);                 // src == dst map (involution)
        float xt = xcb[lsrc];
        float dtx = dt * xt;
        float y = Dv * xt;
        #pragma unroll
        for (int n = 0; n < 16; ++n) {
            h[n] = __expf(dt * A[n]) * h[n] + dtx * sh[16 + n];
            y += h[n] * sh[32 + n];
        }
        atomicAdd(&yb[lsrc], y);
        __syncthreads();
    }
}

// layernorm over 512 + silu(z) gate; one 512-thread block per pixel.
__global__ void gate_ln_kernel(const float* __restrict__ ycomb, const float* __restrict__ xmz,
                               const float* __restrict__ g, const float* __restrict__ be,
                               float* __restrict__ out) {
    __shared__ float red[512];
    int pix = blockIdx.x; int b = pix >> 10, l = pix & 1023;
    int d = threadIdx.x;
    float v = ycomb[((size_t)(b * 512 + d)) * 1024 + l];
    red[d] = v; __syncthreads();
    for (int s = 256; s > 0; s >>= 1) { if (d < s) red[d] += red[d + s]; __syncthreads(); }
    float mu = red[0] * (1.f / 512.f); __syncthreads();
    float dv = v - mu; red[d] = dv * dv; __syncthreads();
    for (int s = 256; s > 0; s >>= 1) { if (d < s) red[d] += red[d + s]; __syncthreads(); }
    float var = red[0] * (1.f / 512.f);
    float y = (dv * rsqrtf(var + 1e-5f)) * g[d] + be[d];
    float z = xmz[(size_t)pix * 1024 + 512 + d];
    out[(size_t)pix * 512 + d] = y * (z / (1.f + __expf(-z)));
}

__global__ void residual1_kernel(const float* __restrict__ x, const float* __restrict__ mod,
                                 const float* __restrict__ ss2d, float* __restrict__ xh) {
    int idx = blockIdx.x * blockDim.x + threadIdx.x;
    if (idx >= 8 * 1024 * 256) return;
    int c = idx & 255; int pix = idx >> 8; int b = pix >> 10, l = pix & 1023;
    xh[idx] = x[((size_t)(b * 256 + c)) * 1024 + l] + mod[b * 1536 + 512 + c] * ss2d[idx];
}

// depthwise 3x3 SAME, no bias, 768 channels, pixel-major in/out (stride 768)
__global__ void dwconv_kernel(const float* __restrict__ in, const float* __restrict__ w,
                              float* __restrict__ out) {
    int idx = blockIdx.x * blockDim.x + threadIdx.x;
    if (idx >= 8 * 1024 * 768) return;
    int c = idx % 768; int pix = idx / 768; int b = pix >> 10, l = pix & 1023;
    int h = l >> 5, ww = l & 31;
    float s = 0.f;
    #pragma unroll
    for (int dh = -1; dh <= 1; ++dh)
        #pragma unroll
        for (int dwi = -1; dwi <= 1; ++dwi) {
            int hh = h + dh, w2 = ww + dwi;
            if (hh < 0 || hh > 31 || w2 < 0 || w2 > 31) continue;
            s += in[((size_t)(b * 1024 + hh * 32 + w2)) * 768 + c]
               * w[c * 9 + (dh + 1) * 3 + (dwi + 1)];
        }
    out[idx] = s;
}

// L2 norms of q,k rows over L=1024. r = b*512 + s*256 + (head*32 + d)
__global__ void qknorm_kernel(const float* __restrict__ dwq, float* __restrict__ nrm) {
    int r = blockIdx.x * blockDim.x + threadIdx.x;
    if (r >= 4096) return;
    int b = r >> 9, rem = r & 511;
    float s = 0.f;
    for (int l = 0; l < 1024; ++l) {
        float v = dwq[((size_t)(b * 1024 + l)) * 768 + rem];
        s += v * v;
    }
    nrm[r] = sqrtf(s);
}

// softmax with cosine scaling * temperature, in-place on G rows of 32
__global__ void softmax_kernel(float* __restrict__ G, const float* __restrict__ nrm,
                               const float* __restrict__ temp) {
    int idx = blockIdx.x * blockDim.x + threadIdx.x;
    if (idx >= 2048) return;
    int cc = idx & 31, bh = idx >> 5; int b = bh >> 3, head = bh & 7;
    float nq = fmaxf(nrm[b * 512 + head * 32 + cc], 1e-12f);
    float tp = temp[head];
    float* row = G + (size_t)bh * 1024 + cc * 32;
    float vals[32]; float mx = -1e30f;
    for (int dd = 0; dd < 32; ++dd) {
        float nk = fmaxf(nrm[b * 512 + 256 + head * 32 + dd], 1e-12f);
        float v = row[dd] / (nq * nk) * tp;
        vals[dd] = v; mx = fmaxf(mx, v);
    }
    float ssum = 0.f;
    for (int dd = 0; dd < 32; ++dd) { vals[dd] = __expf(vals[dd] - mx); ssum += vals[dd]; }
    float inv = 1.f / ssum;
    for (int dd = 0; dd < 32; ++dd) row[dd] = vals[dd] * inv;
}

__global__ void final_kernel(const float* __restrict__ xh, const float* __restrict__ mod,
                             const float* __restrict__ proj, float* __restrict__ out) {
    int idx = blockIdx.x * blockDim.x + threadIdx.x;
    if (idx >= 8 * 1024 * 256) return;
    int c = idx & 255; int pix = idx >> 8; int b = pix >> 10, l = pix & 1023;
    out[((size_t)(b * 256 + c)) * 1024 + l] = xh[idx] + mod[b * 1536 + 5 * 256 + c] * proj[idx];
}

// ---------------------------------------------------------------------------
extern "C" void kernel_launch(void* const* d_in, const int* in_sizes, int n_in,
                              void* d_out, int out_size, void* d_ws, size_t ws_size,
                              hipStream_t stream) {
    const float* x      = (const float*)d_in[0];
    const float* t      = (const float*)d_in[2];
    const float* g1     = (const float*)d_in[3];
    const float* b1     = (const float*)d_in[4];
    const float* Wada   = (const float*)d_in[5];
    const float* bada   = (const float*)d_in[6];
    const float* Win    = (const float*)d_in[7];
    const float* convw  = (const float*)d_in[8];
    const float* convb  = (const float*)d_in[9];
    const float* xpw    = (const float*)d_in[10];
    const float* dtpw   = (const float*)d_in[11];
    const float* dtpb   = (const float*)d_in[12];
    const float* Alogs  = (const float*)d_in[13];
    const float* Ds     = (const float*)d_in[14];
    const float* ong    = (const float*)d_in[15];
    const float* onb    = (const float*)d_in[16];
    const float* Wout   = (const float*)d_in[17];
    const float* temp   = (const float*)d_in[18];
    const float* qkvw   = (const float*)d_in[19];
    const float* dwcw   = (const float*)d_in[20];
    const float* projw  = (const float*)d_in[21];
    float* out = (float*)d_out;

    float* ws = (float*)d_ws;
    // workspace layout (float offsets)
    float* mod     = ws;                 // 12288
    float* qkvWT   = ws + 12288;         // 196608 (256x768)
    float* projWT  = ws + 208896;        // 65536  (256x256)
    float* nrm     = ws + 274432;        // 4096
    float* G       = ws + 278528;        // 65536  (64 x 32 x 32)
    float* h1      = ws + 344064;        // 2097152 (8192x256)
    float* xmz     = ws + 2441216;       // 8388608 (8192x1024)
    float* xc      = ws + 10829824;      // 4194304 (8x512x1024)
    float* xdbl    = ws + 15024128;      // 1572864 (32x48x1024)
    float* ycomb   = ws + 16596992;      // 4194304 (8x512x1024)
    float* ygated  = ws + 20791296;      // 4194304 (8192x512)
    float* ss2d    = ws + 24985600;      // 2097152 (8192x256)
    float* xh      = ws + 27082752;      // 2097152
    float* h2      = ws + 29179904;      // 2097152
    float* qkvpm   = ws + 31277056;      // 6291456 (8192x768)
    float* dwc     = ws + 37568512;      // 6291456
    float* attnout = ws + 43859968;      // 2097152
    float* projout = ws + 45957120;      // 2097152  -> total 48054272 floats

    // 0) zero scan accumulator
    zero_kernel<<<16384, 256, 0, stream>>>(ycomb, 4194304);
    // weight transposes for 1x1 convs (GEMM wants K-major rows)
    transpose_kernel<<<768, 256, 0, stream>>>(qkvw, qkvWT, 768, 256);
    transpose_kernel<<<256, 256, 0, stream>>>(projw, projWT, 256, 256);
    // 1) adaLN modulation vector
    mod_kernel<<<48, 256, 0, stream>>>(t, Wada, bada, mod);
    // 2) h1 = modulate(LN(x), sh_msa, sc_msa)
    ln_mod_kernel<<<8192, 256, 0, stream>>>(x, 1, g1, b1, mod, 0, 1, 1e-5f, h1);
    // 3) in-projection: (8192x256)@(256x1024) -> [xm | z]
    gemm_wmma_kernel<<<dim3(64, 8), 256, 0, stream>>>(h1, Win, xmz, 256, 256, 1024, 1024);
    // 4) depthwise conv + silu -> xc (b,d,l)
    conv_silu_kernel<<<16384, 256, 0, stream>>>(xmz, convw, convb, xc);
    // 5) x_dbl = einsum over d (WMMA, direction remap fused)
    xdbl_wmma_kernel<<<dim3(64, 3, 32), 32, 0, stream>>>(xc, xpw, xdbl);
    // 6) selective scan, 4 directions scatter-added into ycomb
    scan_kernel<<<32, 512, 0, stream>>>(xc, xdbl, dtpw, dtpb, Alogs, Ds, ycomb);
    // 7) out-norm LN * silu(z)
    gate_ln_kernel<<<8192, 512, 0, stream>>>(ycomb, xmz, ong, onb, ygated);
    // 8) out-projection: (8192x512)@(512x256)
    gemm_wmma_kernel<<<dim3(64, 2), 256, 0, stream>>>(ygated, Wout, ss2d, 512, 512, 256, 256);
    // 9) xh = x + g_msa * ss2d
    residual1_kernel<<<8192, 256, 0, stream>>>(x, mod, ss2d, xh);
    // 10) h2 = modulate(LN(xh, eps=1e-6), sh_mlp, sc_mlp)
    ln_mod_kernel<<<8192, 256, 0, stream>>>(xh, 0, nullptr, nullptr, mod, 3, 4, 1e-6f, h2);
    // 11) qkv 1x1: (8192x256)@(256x768)
    gemm_wmma_kernel<<<dim3(64, 6), 256, 0, stream>>>(h2, qkvWT, qkvpm, 256, 256, 768, 768);
    // 12) depthwise 3x3 on qkv
    dwconv_kernel<<<24576, 256, 0, stream>>>(qkvpm, dwcw, dwc);
    // 13) q,k row norms
    qknorm_kernel<<<16, 256, 0, stream>>>(dwc, nrm);
    // 14) Gram matrices (WMMA)
    gram_wmma_kernel<<<64, 128, 0, stream>>>(dwc, G);
    // 15) cosine-scaled softmax (in place)
    softmax_kernel<<<8, 256, 0, stream>>>(G, nrm, temp);
    // 16) attn @ v (WMMA)
    attn_apply_wmma_kernel<<<dim3(64, 64), 64, 0, stream>>>(G, dwc, attnout);
    // 17) proj 1x1: (8192x256)@(256x256)
    gemm_wmma_kernel<<<dim3(64, 2), 256, 0, stream>>>(attnout, projWT, projout, 256, 256, 256, 256);
    // 18) final residual, back to NCHW
    final_kernel<<<8192, 256, 0, stream>>>(xh, mod, projout, out);
}
